// GarenBCPolicy_35450660061495
// MI455X (gfx1250) — compile-verified, matrix operand order checked by reading, not run on hardware
//
#include <hip/hip_runtime.h>
#include <hip/hip_bf16.h>
#include <math.h>

// ---------------------------------------------------------------------------
// GarenBCPolicy forward for MI455X (gfx1250, wave32, WMMA + TDM).
//  * all large GEMMs in bf16 via v_wmma_f32_16x16x32_bf16 (f32 accumulate)
//  * weights pre-transposed to [N][K] bf16 -> A/B fragments are two 16-byte
//    contiguous loads per lane (documented 16-bit operand striping)
//  * dominant GEMM (obs @ Wr1, ~100 GFLOP) uses the Tensor Data Mover to DMA
//    A/B tiles into LDS, double-buffered, overlapped via TENSORcnt
//  * detection MLP layer-1 folded into precomputed U = char_emb@W1+b1
// Workspace use: ~226 MB of d_ws.
// ---------------------------------------------------------------------------

#define DEV __device__ __forceinline__

typedef __attribute__((ext_vector_type(16))) __bf16 v16bf;
typedef __attribute__((ext_vector_type(8)))  __bf16 v8bf;
typedef __attribute__((ext_vector_type(8)))  float  v8f;
typedef __attribute__((ext_vector_type(4)))  unsigned int v4u;
typedef __attribute__((ext_vector_type(8)))  int v8i;
typedef __attribute__((ext_vector_type(4)))  int v4i;

union Frag { v16bf v; v8bf h[2]; };

static constexpr int Bsz = 512, Tsz = 9, BT = Bsz * Tsz;
static constexpr int TC = 65, ED = 128, HALF = 64, NIG = 70, NC = 88;
static constexpr int NS = 24, NM = 24;
static constexpr int OBS  = 21208;      // 88 + 65*256 + 70*64
static constexpr int OBSP = 21216;      // padded to multiple of 32 (K dim)

// Load a 16x32 (A) or 32x16-as-[N][K] (B) bf16 WMMA fragment.
// lane l: row = l&15, kbase = (l>>4)*8; halves at k+0..7 and k+16..23.
DEV v16bf load_frag(const __bf16* base, int ld, int lane) {
  int r  = lane & 15;
  int kb = (lane >> 4) << 3;
  const __bf16* p = base + (size_t)r * ld + kb;
  Frag f;
  f.h[0] = *(const v8bf*)(p);
  f.h[1] = *(const v8bf*)(p + 16);
  return f.v;
}

DEV v8f wmma_bf16(v16bf a, v16bf b, v8f c) {
  return __builtin_amdgcn_wmma_f32_16x16x32_bf16(false, a, false, b,
                                                 (short)0, c, false, false);
}

// --------------------------------------------------------------- TDM helper
// 2D tile load Global->LDS through the Tensor Data Mover (D# per ISA ch.8):
//  group0: count=1 | lds_addr | global_addr[56:0] | type=2
//  group1: data_size=1 (2B), tensor_dim0=kelems, tensor_dim1=rows,
//          tile_dim0=kelems, tile_dim1=rows, tensor_dim0_stride=ld (elems)
//  groups 2/3 (+extra group on this toolchain): zero (2D tensor)
#if __has_builtin(__builtin_amdgcn_tensor_load_to_lds)
#define HAVE_TDM 1
DEV void tdm_load_tile(void* lds_dst, const void* gsrc,
                       int rows, int kelems, int ld_elems) {
  unsigned lds = (unsigned)(uintptr_t)lds_dst;        // LDS aperture low bits
  unsigned long long ga = (unsigned long long)(uintptr_t)gsrc;
  v4u g0 = { 1u,                                       // count=1 (valid D#)
             lds,
             (unsigned)(ga & 0xffffffffu),
             (unsigned)((ga >> 32) & 0x01ffffffull) | (2u << 30) };  // type=2
  unsigned td0 = (unsigned)kelems, td1 = (unsigned)rows;
  unsigned long long st0 = (unsigned long long)ld_elems;
  v8i g1 = { (int)(1u << 16),                                   // data_size=2B
             (int)((td0 & 0xffffu) << 16),                      // tensor_dim0 lo
             (int)(((td0 >> 16) & 0xffffu) | ((td1 & 0xffffu) << 16)),
             (int)(((td1 >> 16) & 0xffffu) | ((td0 & 0xffffu) << 16)), // tile_dim0
             (int)(td1 & 0xffffu),                              // tile_dim1
             (int)(st0 & 0xffffffffu),                          // dim0 stride lo
             (int)((st0 >> 32) & 0xffffu),                      // stride hi
             0 };
  v4i gz4 = { 0, 0, 0, 0 };
  v8i gz8 = { 0, 0, 0, 0, 0, 0, 0, 0 };
  // 6-arg form (clang-23 / therock-10.0 headers): extra int32x8 group + cpol
  __builtin_amdgcn_tensor_load_to_lds(g0, g1, gz4, gz4, gz8, 0);
}
#else
#define HAVE_TDM 0
#endif

// ---------------------------------------------------------------- conversions
__global__ void k_transpose_bf16(__bf16* __restrict__ dst,
                                 const float* __restrict__ src,
                                 int N, int Ksrc, int Kpad) {
  int total = N * Kpad;
  for (int i = blockIdx.x * blockDim.x + threadIdx.x; i < total;
       i += gridDim.x * blockDim.x) {
    int n = i / Kpad, k = i - n * Kpad;
    float v = (k < Ksrc) ? src[(size_t)k * N + n] : 0.f;
    dst[(size_t)n * Kpad + k] = (__bf16)v;
  }
}

// U[c][e] = b1[e] + sum_k char_emb[c][k] * W1[k][e]   (65x128, tiny)
__global__ void k_compute_U(__bf16* __restrict__ U,
                            const float* __restrict__ emb,
                            const float* __restrict__ W1,
                            const float* __restrict__ b1) {
  int i = blockIdx.x * blockDim.x + threadIdx.x;
  if (i >= TC * ED) return;
  int c = i >> 7, e = i & 127;
  float acc = b1[e];
  const float* er = emb + (size_t)c * ED;
  for (int k = 0; k < ED; ++k) acc += er[k] * W1[(size_t)k * ED + e];
  U[i] = (__bf16)acc;
}

// ------------------------------------------------- obs: cont | items | pad
__global__ void __launch_bounds__(256) k_cont_items(
    __bf16* __restrict__ obs, const float* __restrict__ cont,
    const int* __restrict__ items, const float* __restrict__ item_emb) {
  int bt = blockIdx.x;
  __bf16* row = obs + (size_t)bt * OBSP;
  const int NWORK = NC + 8 + NIG * HALF;   // 88 + 8 pad + 4480 item
  for (int i = threadIdx.x; i < NWORK; i += blockDim.x) {
    if (i < NC) {
      row[i] = (__bf16)cont[(size_t)bt * NC + i];
    } else if (i < NC + 8) {
      row[OBS + (i - NC)] = (__bf16)0.f;             // K padding
    } else {
      int idx = i - NC - 8;
      int j = idx >> 6, hh = idx & 63;
      int it = items[(size_t)bt * NIG + j];
      row[NC + TC * 2 * ED + idx] = (__bf16)item_emb[(size_t)it * HALF + hh];
    }
  }
}

// --------------------------------------------- fused detection MLP per (b,t)
__global__ void __launch_bounds__(256) k_det(
    const int* __restrict__ sids, const float* __restrict__ svals,
    const int* __restrict__ mids, const float* __restrict__ mvals,
    const float* __restrict__ Ws1, const float* __restrict__ Wm1,
    const __bf16* __restrict__ Us, const __bf16* __restrict__ Um,
    const __bf16* __restrict__ Ws2t, const __bf16* __restrict__ Wm2t,
    const float* __restrict__ bs2, const float* __restrict__ bm2,
    __bf16* __restrict__ obs) {
  __shared__ float  fs[TC * 4];
  __shared__ float  fm[TC * 2];
  __shared__ __bf16 h1[160 * 128];   // rows 0..79 screen (65 real), 80..159 mini
  int bt = blockIdx.x, tid = threadIdx.x;

  for (int i = tid; i < TC * 4; i += 256) fs[i] = 0.f;
  for (int i = tid; i < TC * 2; i += 256) fm[i] = 0.f;
  __syncthreads();
  // serialized scatter -> deterministic (last detection wins)
  if (tid == 0) {
    for (int j = 0; j < NS; ++j) {
      int id = sids[(size_t)bt * NS + j];
      for (int k = 0; k < 4; ++k)
        fs[id * 4 + k] = svals[((size_t)bt * NS + j) * 4 + k];
    }
  } else if (tid == 32) {
    for (int j = 0; j < NM; ++j) {
      int id = mids[(size_t)bt * NM + j];
      for (int k = 0; k < 2; ++k)
        fm[id * 2 + k] = mvals[((size_t)bt * NM + j) * 2 + k];
    }
  }
  __syncthreads();

  // layer 1: h1 = relu(U[c] + feats . W1_feat_rows), bf16 into LDS
  for (int i = tid; i < 160 * 128; i += 256) {
    int rp = i >> 7, e = i & 127;
    float v = 0.f;
    if (rp < 80) {
      if (rp < TC) {
        v = (float)Us[rp * ED + e]
          + fs[rp * 4 + 0] * Ws1[(ED + 0) * ED + e]
          + fs[rp * 4 + 1] * Ws1[(ED + 1) * ED + e]
          + fs[rp * 4 + 2] * Ws1[(ED + 2) * ED + e]
          + fs[rp * 4 + 3] * Ws1[(ED + 3) * ED + e];
        v = fmaxf(v, 0.f);
      }
    } else {
      int c = rp - 80;
      if (c < TC) {
        v = (float)Um[c * ED + e]
          + fm[c * 2 + 0] * Wm1[(ED + 0) * ED + e]
          + fm[c * 2 + 1] * Wm1[(ED + 1) * ED + e];
        v = fmaxf(v, 0.f);
      }
    }
    h1[rp * 128 + e] = (__bf16)v;
  }
  __syncthreads();

  // layer 2: [160x128] @ [128x128] via WMMA, write straight into obs row
  int wave = tid >> 5, lane = tid & 31;
  __bf16* row = obs + (size_t)bt * OBSP;
  for (int t = wave; t < 80; t += 8) {      // 10 M-tiles x 8 N-tiles
    int mt = t >> 3, nt = t & 7;
    bool scr = (mt < 5);
    const __bf16* W2t = scr ? Ws2t : Wm2t;
    const float*  b2  = scr ? bs2  : bm2;
    v8f acc = {0.f, 0.f, 0.f, 0.f, 0.f, 0.f, 0.f, 0.f};
    for (int kk = 0; kk < 128; kk += 32) {
      v16bf a = load_frag(&h1[(mt * 16) * 128 + kk], 128, lane);
      v16bf b = load_frag(&W2t[(size_t)(nt * 16) * 128 + kk], 128, lane);
      acc = wmma_bf16(a, b, acc);
    }
    int n = nt * 16 + (lane & 15);
    float bias = b2[n];
    int mrow = mt * 16 + ((lane >> 4) << 3);
    for (int i = 0; i < 8; ++i) {
      int m = mrow + i;
      int c = scr ? m : (m - 80);
      if (c < TC) {
        int col = NC + c * 256 + (scr ? 0 : 128) + n;
        row[col] = (__bf16)(acc[i] + bias);
      }
    }
  }
}

// ---------------- TDM-fed, double-buffered bf16 WMMA GEMM + bias + relu
// C[M x N](bf16) = relu(A[M x K](bf16, ldA) @ Bt[N x K]^T (ldB) + bias[N])
// block tile 64x128, wave tile 32x32, K multiple of 32.
// wave0 DMAs the B tile, wave1 the A tile; TENSORcnt<=1 keeps one DMA in
// flight while WMMA consumes the other buffer.
__global__ void __launch_bounds__(256) k_gemm_relu(
    const __bf16* __restrict__ A, int ldA,
    const __bf16* __restrict__ Bt, int ldB,
    const float* __restrict__ bias,
    __bf16* __restrict__ C, int ldC, int K) {
  __shared__ __bf16 As[2][64 * 32];
  __shared__ __bf16 Bs[2][128 * 32];
  int mb = blockIdx.x, nb = blockIdx.y;
  int tid = threadIdx.x, wave = tid >> 5, lane = tid & 31;
  int wm = wave >> 2, wn = wave & 3;
  v8f z = {0.f, 0.f, 0.f, 0.f, 0.f, 0.f, 0.f, 0.f};
  v8f acc[2][2] = {{z, z}, {z, z}};
  const __bf16* Ab = A  + (size_t)(mb * 64)  * ldA;
  const __bf16* Bb = Bt + (size_t)(nb * 128) * ldB;
  int nIter = K >> 5;
#if HAVE_TDM
  if (wave == 0) tdm_load_tile(Bs[0], Bb, 128, 32, ldB);
  if (wave == 1) tdm_load_tile(As[0], Ab, 64, 32, ldA);
  for (int it = 0; it < nIter; ++it) {
    int cur = it & 1, nxt = cur ^ 1;
    bool more = (it + 1 < nIter);
    if (more) {   // kick DMA for next tile into the other buffer
      if (wave == 0) tdm_load_tile(Bs[nxt], Bb + (size_t)(it + 1) * 32, 128, 32, ldB);
      if (wave == 1) tdm_load_tile(As[nxt], Ab + (size_t)(it + 1) * 32, 64, 32, ldA);
    }
    if (wave < 2) {   // issuing waves retire the older DMA, then publish
      if (more) __builtin_amdgcn_s_wait_tensorcnt((short)1);
      else      __builtin_amdgcn_s_wait_tensorcnt((short)0);
    }
    __syncthreads();
    v16bf a0 = load_frag(&As[cur][(wm * 32 +  0) * 32], 32, lane);
    v16bf a1 = load_frag(&As[cur][(wm * 32 + 16) * 32], 32, lane);
    v16bf b0 = load_frag(&Bs[cur][(wn * 32 +  0) * 32], 32, lane);
    v16bf b1 = load_frag(&Bs[cur][(wn * 32 + 16) * 32], 32, lane);
    acc[0][0] = wmma_bf16(a0, b0, acc[0][0]);
    acc[0][1] = wmma_bf16(a0, b1, acc[0][1]);
    acc[1][0] = wmma_bf16(a1, b0, acc[1][0]);
    acc[1][1] = wmma_bf16(a1, b1, acc[1][1]);
    __syncthreads();  // reads done before this buffer is re-targeted
  }
#else
  int r = tid >> 2, s = tid & 3;
  for (int it = 0; it < nIter; ++it) {
    int k0 = it * 32;
    *(v8bf*)&As[0][r * 32 + s * 8] = *(const v8bf*)&Ab[(size_t)r * ldA + k0 + s * 8];
    *(v8bf*)&Bs[0][r * 32 + s * 8] = *(const v8bf*)&Bb[(size_t)r * ldB + k0 + s * 8];
    *(v8bf*)&Bs[0][(r + 64) * 32 + s * 8] =
        *(const v8bf*)&Bb[(size_t)(r + 64) * ldB + k0 + s * 8];
    __syncthreads();
    v16bf a0 = load_frag(&As[0][(wm * 32 +  0) * 32], 32, lane);
    v16bf a1 = load_frag(&As[0][(wm * 32 + 16) * 32], 32, lane);
    v16bf b0 = load_frag(&Bs[0][(wn * 32 +  0) * 32], 32, lane);
    v16bf b1 = load_frag(&Bs[0][(wn * 32 + 16) * 32], 32, lane);
    acc[0][0] = wmma_bf16(a0, b0, acc[0][0]);
    acc[0][1] = wmma_bf16(a0, b1, acc[0][1]);
    acc[1][0] = wmma_bf16(a1, b0, acc[1][0]);
    acc[1][1] = wmma_bf16(a1, b1, acc[1][1]);
    __syncthreads();
  }
#endif
  for (int ti = 0; ti < 2; ++ti)
    for (int tj = 0; tj < 2; ++tj) {
      int n = nb * 128 + wn * 32 + tj * 16 + (lane & 15);
      float bv = bias[n];
      int mbase = mb * 64 + wm * 32 + ti * 16 + ((lane >> 4) << 3);
      for (int i = 0; i < 8; ++i) {
        float v = fmaxf(acc[ti][tj][i] + bv, 0.f);
        C[(size_t)(mbase + i) * ldC + n] = (__bf16)v;
      }
    }
}

// ----------------------- LSTM gate GEMM: g = x_t@Wi + h@Wh + bi + bh (f32 out)
__global__ void __launch_bounds__(256) k_gates(
    const __bf16* __restrict__ X, int ldX,      // rows b at X + b*ldX, K=128
    const __bf16* __restrict__ Wit,             // [512][128] bf16
    const __bf16* __restrict__ Hb,              // [512 x 128] bf16
    const __bf16* __restrict__ Wht,             // [512][128] bf16
    const float* __restrict__ bi, const float* __restrict__ bh,
    float* __restrict__ g) {                    // [512 x 512] f32
  __shared__ __bf16 As[64 * 32];
  __shared__ __bf16 Bs[128 * 32];
  int mb = blockIdx.x, nb = blockIdx.y;
  int tid = threadIdx.x, wave = tid >> 5, lane = tid & 31;
  int wm = wave >> 2, wn = wave & 3;
  v8f z = {0.f, 0.f, 0.f, 0.f, 0.f, 0.f, 0.f, 0.f};
  v8f acc[2][2] = {{z, z}, {z, z}};
  int r = tid >> 2, s = tid & 3;
  for (int phase = 0; phase < 2; ++phase) {
    const __bf16* Ap  = phase ? Hb  : X;
    int           lA  = phase ? 128 : ldX;
    const __bf16* Btp = phase ? Wht : Wit;
    const __bf16* Ab = Ap  + (size_t)(mb * 64)  * lA;
    const __bf16* Bb = Btp + (size_t)(nb * 128) * 128;
    for (int k0 = 0; k0 < 128; k0 += 32) {
      *(v8bf*)&As[r * 32 + s * 8] = *(const v8bf*)&Ab[(size_t)r * lA + k0 + s * 8];
      *(v8bf*)&Bs[r * 32 + s * 8] = *(const v8bf*)&Bb[(size_t)r * 128 + k0 + s * 8];
      *(v8bf*)&Bs[(r + 64) * 32 + s * 8] =
          *(const v8bf*)&Bb[(size_t)(r + 64) * 128 + k0 + s * 8];
      if (k0 + 32 < 128)
        __builtin_prefetch(&Ab[(size_t)r * lA + k0 + 32], 0, 0);
      __syncthreads();
      v16bf a0 = load_frag(&As[(wm * 32 +  0) * 32], 32, lane);
      v16bf a1 = load_frag(&As[(wm * 32 + 16) * 32], 32, lane);
      v16bf b0 = load_frag(&Bs[(wn * 32 +  0) * 32], 32, lane);
      v16bf b1 = load_frag(&Bs[(wn * 32 + 16) * 32], 32, lane);
      acc[0][0] = wmma_bf16(a0, b0, acc[0][0]);
      acc[0][1] = wmma_bf16(a0, b1, acc[0][1]);
      acc[1][0] = wmma_bf16(a1, b0, acc[1][0]);
      acc[1][1] = wmma_bf16(a1, b1, acc[1][1]);
      __syncthreads();
    }
  }
  for (int ti = 0; ti < 2; ++ti)
    for (int tj = 0; tj < 2; ++tj) {
      int n = nb * 128 + wn * 32 + tj * 16 + (lane & 15);
      float bv = bi[n] + bh[n];
      int mbase = mb * 64 + wm * 32 + ti * 16 + ((lane >> 4) << 3);
      for (int i = 0; i < 8; ++i)
        g[(size_t)(mbase + i) * 512 + n] = acc[ti][tj][i] + bv;
    }
}

// --------------------------------------------------------- LSTM elementwise
__global__ void k_lstm_elem(const float* __restrict__ g,
                            float* __restrict__ h, float* __restrict__ c,
                            __bf16* __restrict__ hb) {
  int i = blockIdx.x * blockDim.x + threadIdx.x;
  if (i >= Bsz * ED) return;
  int b = i >> 7, e = i & 127;
  const float* gr = g + (size_t)b * 512;
  float ig = gr[e], fg = gr[128 + e], gg = gr[256 + e], og = gr[384 + e];
  float si = 1.f / (1.f + __expf(-ig));
  float sf = 1.f / (1.f + __expf(-fg));
  float so = 1.f / (1.f + __expf(-og));
  float cc = sf * c[i] + si * tanhf(gg);
  float hh = so * tanhf(cc);
  c[i] = cc;
  h[i] = hh;
  hb[i] = (__bf16)hh;
}

__global__ void k_init_state(float* __restrict__ h, float* __restrict__ c,
                             __bf16* __restrict__ hb) {
  int i = blockIdx.x * blockDim.x + threadIdx.x;
  if (i >= Bsz * ED) return;
  h[i] = 0.f; c[i] = 0.f; hb[i] = (__bf16)0.f;
}

// ------------------------------------------------------------------- heads
__global__ void k_heads(const float* __restrict__ h,
                        const float* __restrict__ Wm, const float* __restrict__ bm,
                        const float* __restrict__ Wa, const float* __restrict__ ba,
                        const float* __restrict__ Wt, const float* __restrict__ btg,
                        const float* __restrict__ Wab, const float* __restrict__ bab,
                        float* __restrict__ out) {
  int i = blockIdx.x * blockDim.x + threadIdx.x;
  if (i >= Bsz * 34) return;
  int b = i / 34, col = i - b * 34;
  const float* hr = h + (size_t)b * ED;
  const float* W; const float* bb; int nc, cc; bool sg = false;
  if (col < 25)      { W = Wm;  bb = bm;  nc = 25; cc = col; }
  else if (col < 26) { W = Wa;  bb = ba;  nc = 1;  cc = 0; }
  else if (col < 28) { W = Wt;  bb = btg; nc = 2;  cc = col - 26; sg = true; }
  else               { W = Wab; bb = bab; nc = 6;  cc = col - 28; }
  float acc = bb[cc];
  for (int k = 0; k < ED; ++k) acc += hr[k] * W[(size_t)k * nc + cc];
  if (sg) acc = 1.f / (1.f + __expf(-acc));
  out[i] = acc;
}

// ---------------------------------------------------------------------------
extern "C" void kernel_launch(void* const* d_in, const int* in_sizes, int n_in,
                              void* d_out, int out_size, void* d_ws,
                              size_t ws_size, hipStream_t stream) {
  const float* cont     = (const float*)d_in[0];
  const int*   sids     = (const int*)d_in[1];
  const float* svals    = (const float*)d_in[2];
  const int*   mids     = (const int*)d_in[3];
  const float* mvals    = (const float*)d_in[4];
  const int*   items    = (const int*)d_in[5];
  const float* char_emb = (const float*)d_in[6];
  const float* item_emb = (const float*)d_in[7];
  const float* Ws1 = (const float*)d_in[8],  *bs1 = (const float*)d_in[9];
  const float* Ws2 = (const float*)d_in[10], *bs2 = (const float*)d_in[11];
  const float* Wm1 = (const float*)d_in[12], *bm1 = (const float*)d_in[13];
  const float* Wm2 = (const float*)d_in[14], *bm2 = (const float*)d_in[15];
  const float* Wr1 = (const float*)d_in[16], *br1 = (const float*)d_in[17];
  const float* Wr2 = (const float*)d_in[18], *br2 = (const float*)d_in[19];
  const float* Wi  = (const float*)d_in[20], *Wh  = (const float*)d_in[21];
  const float* bi  = (const float*)d_in[22], *bh  = (const float*)d_in[23];
  const float* Wmv = (const float*)d_in[24], *bmv = (const float*)d_in[25];
  const float* Wak = (const float*)d_in[26], *bak = (const float*)d_in[27];
  const float* Wtg = (const float*)d_in[28], *btg = (const float*)d_in[29];
  const float* Wab = (const float*)d_in[30], *bab = (const float*)d_in[31];
  float* out = (float*)d_out;

  char* ws = (char*)d_ws;
  size_t off = 0;
  auto alloc = [&](size_t bytes) -> void* {
    void* p = ws + off;
    off = (off + bytes + 255) & ~(size_t)255;
    return p;
  };
  __bf16* obs  = (__bf16*)alloc((size_t)BT * OBSP * 2);   // ~195.5 MB
  __bf16* Wr1t = (__bf16*)alloc((size_t)512 * OBSP * 2);  // ~21.7 MB
  __bf16* Wr2t = (__bf16*)alloc((size_t)128 * 512 * 2);
  __bf16* Ws2t = (__bf16*)alloc((size_t)128 * 128 * 2);
  __bf16* Wm2t = (__bf16*)alloc((size_t)128 * 128 * 2);
  __bf16* Wit  = (__bf16*)alloc((size_t)512 * 128 * 2);
  __bf16* Wht  = (__bf16*)alloc((size_t)512 * 128 * 2);
  __bf16* Us   = (__bf16*)alloc((size_t)TC * ED * 2);
  __bf16* Um   = (__bf16*)alloc((size_t)TC * ED * 2);
  __bf16* x1b  = (__bf16*)alloc((size_t)BT * 512 * 2);
  __bf16* xb   = (__bf16*)alloc((size_t)BT * 128 * 2);
  float*  g    = (float*)alloc((size_t)Bsz * 512 * 4);
  float*  hs   = (float*)alloc((size_t)Bsz * ED * 4);
  float*  cs   = (float*)alloc((size_t)Bsz * ED * 4);
  __bf16* hb   = (__bf16*)alloc((size_t)Bsz * ED * 2);
  (void)ws_size; (void)in_sizes; (void)n_in; (void)out_size;

  dim3 blk(256);
  // weight conversions (bf16, [N][K] transposed, K zero-padded)
  k_transpose_bf16<<<2048, blk, 0, stream>>>(Wr1t, Wr1, 512, OBS, OBSP);
  k_transpose_bf16<<<256,  blk, 0, stream>>>(Wr2t, Wr2, 128, 512, 512);
  k_transpose_bf16<<<64,   blk, 0, stream>>>(Ws2t, Ws2, 128, 128, 128);
  k_transpose_bf16<<<64,   blk, 0, stream>>>(Wm2t, Wm2, 128, 128, 128);
  k_transpose_bf16<<<256,  blk, 0, stream>>>(Wit,  Wi,  512, 128, 128);
  k_transpose_bf16<<<256,  blk, 0, stream>>>(Wht,  Wh,  512, 128, 128);
  k_compute_U<<<(TC * ED + 255) / 256, blk, 0, stream>>>(Us, char_emb, Ws1, bs1);
  k_compute_U<<<(TC * ED + 255) / 256, blk, 0, stream>>>(Um, char_emb, Wm1, bm1);

  // obs assembly
  k_cont_items<<<BT, blk, 0, stream>>>(obs, cont, items, item_emb);
  k_det<<<BT, blk, 0, stream>>>(sids, svals, mids, mvals, Ws1, Wm1,
                                Us, Um, Ws2t, Wm2t, bs2, bm2, obs);

  // reducer MLP: x1 = relu(obs@Wr1+br1); x = relu(x1@Wr2+br2)
  k_gemm_relu<<<dim3(BT / 64, 4), blk, 0, stream>>>(obs, OBSP, Wr1t, OBSP,
                                                    br1, x1b, 512, OBSP);
  k_gemm_relu<<<dim3(BT / 64, 1), blk, 0, stream>>>(x1b, 512, Wr2t, 512,
                                                    br2, xb, 128, 512);

  // LSTM over T=9
  k_init_state<<<(Bsz * ED + 255) / 256, blk, 0, stream>>>(hs, cs, hb);
  for (int t = 0; t < Tsz; ++t) {
    k_gates<<<dim3(Bsz / 64, 4), blk, 0, stream>>>(xb + (size_t)t * ED,
                                                   Tsz * ED, Wit, hb, Wht,
                                                   bi, bh, g);
    k_lstm_elem<<<(Bsz * ED + 255) / 256, blk, 0, stream>>>(g, hs, cs, hb);
  }

  // heads -> out[512 x 34]
  k_heads<<<(Bsz * 34 + 255) / 256, blk, 0, stream>>>(hs, Wmv, bmv, Wak, bak,
                                                      Wtg, btg, Wab, bab, out);
}
